// DDPM_40389872451639
// MI455X (gfx1250) — matrix operand order repaired and checked by reading, not compiled, not run on hardware
//
#include <hip/hip_runtime.h>
#include <hip/hip_bf16.h>

typedef __attribute__((ext_vector_type(16))) _Float16 v16h;
typedef __attribute__((ext_vector_type(8)))  _Float16 v8h;
typedef __attribute__((ext_vector_type(8)))  float    v8f;

#define NN 4
#define CC 64
#define HH 96
#define WW 96
#define TW 6                 // W/16 tiles per row
#define NBLK (NN*HH*TW)      // 2304
// ws layout (bytes)
#define R_BYTES   (3u*NN*CC*HH*WW*4u)          // 28,311,552
#define GWH_OFF   R_BYTES
#define GWH_BYTES (1728u*64u*2u)               // 221,184
#define FWH_OFF   (GWH_OFF + GWH_BYTES)        // 28,532,736

#define YPITCH  72           // LDS row pitch (halves), mult of 8 -> 16B-aligned rows
#define CPITCH  264          // LDS cat pitch (halves), mult of 8 -> 16B-aligned rows

// Combine two 8-half LDS vectors into one 16-half WMMA A fragment.
__device__ __forceinline__ v16h make_a(const _Float16* row, int base) {
    v8h lo = *(const v8h*)(row + base);
    v8h hi = *(const v8h*)(row + base + 8);
    return __builtin_shufflevector(lo, hi, 0, 1, 2, 3, 4, 5, 6, 7,
                                           8, 9, 10, 11, 12, 13, 14, 15);
}

// ---------------- prep: f32 -> f16 weight reshapes ----------------
__global__ void DDPM_prep_kernel(const float* __restrict__ gen_w,
                                 const float* __restrict__ fuse_w,
                                 _Float16* __restrict__ gwh,
                                 _Float16* __restrict__ fwh) {
    int idx = blockIdx.x * 256 + threadIdx.x;
    const int NG = 1728 * 64;
    if (idx < NG) {
        gwh[idx] = (_Float16)gen_w[idx];
    } else {
        int t = idx - NG;                      // [(i*3+j)*64 + co]*256 + ci
        if (t < 9 * 64 * 256) {
            int ci = t & 255;
            int rest = t >> 8;
            int co = rest & 63;
            int ij = rest >> 6;                // 0..8
            int i = ij / 3, j = ij % 3;
            fwh[t] = (_Float16)fuse_w[((co * 256 + ci) * 3 + i) * 3 + j];
        }
    }
}

// ---------------- kernel-gen GEMM (WMMA) + dynamic convs ----------------
__global__ __launch_bounds__(128) void DDPM_genconv_kernel(
    const float* __restrict__ x, const float* __restrict__ y,
    const _Float16* __restrict__ gwh, const float* __restrict__ gen_b,
    float* __restrict__ r /* [3][N][C][H][W] */) {
    __shared__ __align__(16) _Float16 yT[16 * YPITCH];   // [pixel][c]
    __shared__ __align__(16) float kern[576 * 16];       // [o_local][pixel], one branch

    const int tid  = threadIdx.x;
    const int lane = tid & 31;
    const int wave = tid >> 5;
    const int blk  = blockIdx.x;
    const int tw = blk % TW;
    const int h  = (blk / TW) % HH;
    const int n  = blk / (TW * HH);
    const int w0 = tw * 16;

    // stage y tile [16 px][64 c] as f16 (coalesced along w)
    #pragma unroll
    for (int s = 0; s < 8; ++s) {
        int flat = tid + s * 128;              // 1024 elements
        int c = flat >> 4, p = flat & 15;
        yT[p * YPITCH + c] = (_Float16)y[((n * CC + c) * HH + h) * WW + w0 + p];
    }
    __syncthreads();

    // A fragments (16x32 f16): per lane, halves 0..15 are one contiguous
    // 32B run at K-base = 16*(lane/16) (+32 for second k-block).
    const int m  = lane & 15;
    const int hw = lane >> 4;
    const _Float16* yrow = yT + m * YPITCH;
    v16h a0 = make_a(yrow, hw * 16);
    v16h a1 = make_a(yrow, hw * 16 + 32);

    const int dils[3] = {1, 3, 5};
    for (int di = 0; di < 3; ++di) {
        __syncthreads();                       // kern reuse fence
        // 36 o-tiles of 16; each wave does 9
        for (int it = 0; it < 9; ++it) {
            int otl = wave * 9 + it;           // local o-tile
            int olocal = otl * 16 + m;
            int o = di * 576 + olocal;
            // B fragment (32x16 f16): N=o, K=c ; contiguous 32B loads
            v16h b0 = *(const v16h*)(gwh + o * 64 + hw * 16);
            v16h b1 = *(const v16h*)(gwh + o * 64 + 32 + hw * 16);
            v8f acc = {};
            acc = __builtin_amdgcn_wmma_f32_16x16x32_f16(false, a0, false, b0,
                                                         (short)0, acc, false, false);
            acc = __builtin_amdgcn_wmma_f32_16x16x32_f16(false, a1, false, b1,
                                                         (short)0, acc, false, false);
            float gb = gen_b[o];
            #pragma unroll
            for (int rI = 0; rI < 8; ++rI) {
                int pixel = rI + 8 * hw;       // C/D layout: M = r + 8*(lane/16)
                kern[olocal * 16 + pixel] = acc[rI] + gb;
            }
        }
        __syncthreads();
        // dynamic depthwise conv for this dilation, kernels from LDS
        int d = dils[di];
        #pragma unroll
        for (int s = 0; s < 8; ++s) {
            int vi = tid + s * 128;            // 64c * 16px
            int c = vi >> 4, p = vi & 15;
            const float* xb = x + ((size_t)(n * CC + c) * HH) * WW;
            const float* kp = kern + (c * 9) * 16 + p;
            float acc2 = 0.f;
            #pragma unroll
            for (int t = 0; t < 9; ++t) {
                int i = t / 3, j = t % 3;
                int hh = h + (i - 1) * d;
                int ww = w0 + p + (j - 1) * d;
                float xv = (hh >= 0 && hh < HH && ww >= 0 && ww < WW)
                               ? xb[hh * WW + ww] : 0.f;
                acc2 += xv * kp[t * 16];
            }
            r[((((size_t)di * NN + n) * CC + c) * HH + h) * WW + w0 + p] = acc2;
        }
    }
}

// ---------------- fuse conv 3x3, 256->64 (implicit GEMM, WMMA) ----------------
__global__ __launch_bounds__(128) void DDPM_fuse_kernel(
    const float* __restrict__ x, const float* __restrict__ r,
    const _Float16* __restrict__ fwh, const float* __restrict__ fuse_b,
    float* __restrict__ out) {
    __shared__ __align__(16) _Float16 cat[3 * 18 * CPITCH];  // [row][col][ci]

    const int tid  = threadIdx.x;
    const int lane = tid & 31;
    const int wave = tid >> 5;                 // co-tile
    const int blk  = blockIdx.x;
    const int tw = blk % TW;
    const int h  = (blk / TW) % HH;
    const int n  = blk / (TW * HH);
    const int w0 = tw * 16;

    // stage cat halo tile: 256 ci x 3 rows x 18 cols
    for (int s = 0; s < 108; ++s) {
        int flat = s * 128 + tid;              // 13824 elements
        int ci = flat / 54;
        int rem = flat % 54;
        int i = rem / 18, jj = rem % 18;
        int hh = h + i - 1, ww = w0 + jj - 1;
        float val = 0.f;
        if (hh >= 0 && hh < HH && ww >= 0 && ww < WW) {
            int b = ci >> 6, c = ci & 63;
            const float* src = (b == 0) ? x : (r + (size_t)(b - 1) * NN * CC * HH * WW);
            val = src[((size_t)(n * CC + c) * HH + hh) * WW + ww];
        }
        cat[(i * 18 + jj) * CPITCH + ci] = (_Float16)val;
    }
    __syncthreads();

    const int m   = lane & 15;
    const int hw2 = lane >> 4;
    const int co  = wave * 16 + m;
    v8f acc = {};
    for (int ij = 0; ij < 9; ++ij) {
        int i = ij / 3, j = ij % 3;
        int col = m + j;                       // 0..17
        const _Float16* arow = cat + (i * 18 + col) * CPITCH;
        const _Float16* brow = fwh + ((size_t)ij * 64 + co) * 256 + hw2 * 16;
        #pragma unroll
        for (int kk = 0; kk < 8; ++kk) {       // K = 256 in blocks of 32
            v16h a = make_a(arow, kk * 32 + hw2 * 16);   // 2x ds_load_b128
            v16h b = *(const v16h*)(brow + kk * 32);     // 2x global_load_b128
            acc = __builtin_amdgcn_wmma_f32_16x16x32_f16(false, a, false, b,
                                                         (short)0, acc, false, false);
        }
    }
    float fb = fuse_b[co];
    #pragma unroll
    for (int rI = 0; rI < 8; ++rI) {
        int pixel = rI + 8 * hw2;
        out[((size_t)(n * CC + co) * HH + h) * WW + w0 + pixel] = acc[rI] + fb;
    }
}

extern "C" void kernel_launch(void* const* d_in, const int* in_sizes, int n_in,
                              void* d_out, int out_size, void* d_ws, size_t ws_size,
                              hipStream_t stream) {
    (void)in_sizes; (void)n_in; (void)out_size; (void)ws_size;
    const float* x      = (const float*)d_in[0];
    const float* y      = (const float*)d_in[1];
    const float* gen_w  = (const float*)d_in[2];
    const float* gen_b  = (const float*)d_in[3];
    const float* fuse_w = (const float*)d_in[4];
    const float* fuse_b = (const float*)d_in[5];
    float* out = (float*)d_out;

    char* ws = (char*)d_ws;
    float*    rbuf = (float*)ws;
    _Float16* gwh  = (_Float16*)(ws + GWH_OFF);
    _Float16* fwh  = (_Float16*)(ws + FWH_OFF);

    int prep_total = 1728 * 64 + 9 * 64 * 256;
    DDPM_prep_kernel<<<(prep_total + 255) / 256, 256, 0, stream>>>(gen_w, fuse_w, gwh, fwh);
    DDPM_genconv_kernel<<<NBLK, 128, 0, stream>>>(x, y, gwh, gen_b, rbuf);
    DDPM_fuse_kernel<<<NBLK, 128, 0, stream>>>(x, rbuf, fwh, fuse_b, out);
}
